// MyGCNNModel_66580583022838
// MI455X (gfx1250) — matrix operand order repaired and checked by reading, not compiled, not run on hardware
//
#include <hip/hip_runtime.h>

typedef __attribute__((ext_vector_type(16))) _Float16 v16h;
typedef __attribute__((ext_vector_type(8)))  float    v8f;
typedef __attribute__((ext_vector_type(4)))  float    fx4;
typedef __attribute__((ext_vector_type(2)))  float    fx2;

#define NN  18
#define TPB 128   // items per block, 4 waves of 32

__global__ __launch_bounds__(TPB) void gcn_mlp_fused(
    const float* __restrict__ nf,     // [B,18,1]
    const float* __restrict__ adj,    // [B,18,18]
    const float* __restrict__ proj_w, // [1,1]
    const float* __restrict__ proj_b, // [1]
    const float* __restrict__ fl1_w,  // [10,18]
    const float* __restrict__ fl1_b,  // [10]
    const float* __restrict__ fl2_w,  // [8,10]
    const float* __restrict__ fl2_b,  // [8]
    const float* __restrict__ fl3_w,  // [2,8]
    const float* __restrict__ fl3_b,  // [2]
    float* __restrict__ out,          // [B,2]
    int B)
{
    __shared__ _Float16 lx[TPB][32];  // f16 GCN outputs, K padded 18->32 (WMMA A staging)
    __shared__ float    ly[TPB][12];  // fl1 outputs (10 used, padded)

    const int  tid    = threadIdx.x;
    const int  b      = blockIdx.x * TPB + tid;
    const bool active = (b < B);
    const int  bb     = active ? b : (B - 1);   // clamp: keep EXEC full, no per-load guards

    const float pw = proj_w[0];
    const float pb = proj_b[0];

    // ---------------- per-graph GCN (2 layers), two passes over adj ----------------
    float x[NN];
    float rdeg[NN];
    {
        const float* __restrict__ a0 = adj + (size_t)bb * (NN * NN);
        const float* __restrict__ f0 = nf  + (size_t)bb * NN;

        float h1[NN];
        #pragma unroll
        for (int m = 0; m < NN; m += 2) {
            fx2 v = *(const fx2*)(f0 + m);
            h1[m] = v.x; h1[m + 1] = v.y;
        }

        // pass 1: two rows (144 B = 9 x float4, 16B-aligned) per step.
        // Fused row-sum (degree) + adj @ h1 ; y = pw*dot*rdeg + pb ; relu.
        #pragma unroll
        for (int p = 0; p < NN / 2; ++p) {
            fx4 q[9];
            #pragma unroll
            for (int j = 0; j < 9; ++j)
                q[j] = *(const fx4*)(a0 + p * 36 + j * 4);
            float t[36];
            #pragma unroll
            for (int j = 0; j < 9; ++j) {
                t[4*j] = q[j].x; t[4*j+1] = q[j].y; t[4*j+2] = q[j].z; t[4*j+3] = q[j].w;
            }
            #pragma unroll
            for (int r = 0; r < 2; ++r) {
                const int n = 2 * p + r;
                float s = 0.f, d = 0.f;
                #pragma unroll
                for (int m = 0; m < NN; ++m) {
                    float a = t[18 * r + m];
                    s += a;
                    d += a * h1[m];
                }
                float rs = __builtin_amdgcn_rcpf(s);   // 1-ulp reciprocal
                rdeg[n] = rs;
                x[n] = fmaxf(pw * d * rs + pb, 0.f);
            }
        }

        // pass 2: same stream, warm in WGP$/L2, dead afterwards -> nontemporal
        float y[NN];
        #pragma unroll
        for (int p = 0; p < NN / 2; ++p) {
            fx4 q[9];
            #pragma unroll
            for (int j = 0; j < 9; ++j)
                q[j] = __builtin_nontemporal_load((const fx4*)(a0 + p * 36 + j * 4));
            float t[36];
            #pragma unroll
            for (int j = 0; j < 9; ++j) {
                t[4*j] = q[j].x; t[4*j+1] = q[j].y; t[4*j+2] = q[j].z; t[4*j+3] = q[j].w;
            }
            #pragma unroll
            for (int r = 0; r < 2; ++r) {
                const int n = 2 * p + r;
                float d = 0.f;
                #pragma unroll
                for (int m = 0; m < NN; ++m)
                    d += t[18 * r + m] * x[m];
                y[n] = fmaxf(pw * d * rdeg[n] + pb, 0.f);
            }
        }

        // NaN scrub (matches reference position: after GCN stack)
        #pragma unroll
        for (int n = 0; n < NN; ++n) {
            float v = y[n];
            x[n] = (v != v) ? 0.f : v;
        }
    }

    // stage to LDS as f16, zero-pad K to 32
    #pragma unroll
    for (int k = 0; k < NN; ++k) lx[tid][k] = (_Float16)x[k];
    #pragma unroll
    for (int k = NN; k < 32; ++k) lx[tid][k] = (_Float16)0.f;
    __syncthreads();

    // ---------------- fl1: v_wmma_f32_16x16x32_f16, two 16-row tiles per wave ----------------
    const int  wave = tid >> 5;
    const int  lane = tid & 31;
    const int  lo16 = lane & 15;
    const bool hi   = (lane >= 16);

    // B fragment (32x16 f16): lane%16 = N; elems 0-7 hold K 0-7 (lo) / 8-15 (hi), elems 8-15 hold K 16-23 (lo) / 24-31 (hi).
    // B[k][n] = fl1_w[n*18 + k] for n<10, k<18, else 0.
    v16h bfrag;
    #pragma unroll
    for (int j = 0; j < 16; ++j) {
        int k = (j < 8) ? ((hi ? 8 : 0) + j) : (16 + (hi ? 8 : 0) + (j - 8));
        bfrag[j] = (k < NN && lo16 < 10) ? (_Float16)fl1_w[lo16 * NN + k]
                                         : (_Float16)0.f;
    }
    const float bias1 = (lo16 < 10) ? fl1_b[lo16] : 0.f;

    #pragma unroll
    for (int t = 0; t < 2; ++t) {
        const int R    = wave * 32 + t * 16;  // tile base row (block-local item)
        const int arow = R + lo16;

        // A fragment (16x32 f16): lane%16 = M; VGPR0-3 cover K 0-7 (lo) / 8-15 (hi),
        // VGPR4-7 cover K 16-23 (lo) / 24-31 (hi).
        v16h afrag;
        #pragma unroll
        for (int j = 0; j < 16; ++j) {
            int k = (j < 8) ? ((hi ? 8 : 0) + j)
                            : (16 + (hi ? 8 : 0) + (j - 8));
            afrag[j] = lx[arow][k];
        }

        v8f acc = {};
        acc = __builtin_amdgcn_wmma_f32_16x16x32_f16(
                  false, afrag, false, bfrag, (short)0, acc, false, false);

        // D layout: VGPR r = row M=r (lanes 0-15) / M=r+8 (lanes 16-31), lane%16 = N
        #pragma unroll
        for (int r = 0; r < 8; ++r) {
            int row = R + r + (hi ? 8 : 0);
            if (lo16 < 10)
                ly[row][lo16] = fmaxf(acc[r] + bias1, 0.f);
        }
    }
    __syncthreads();

    // ---------------- fl2 (10->8) + fl3 (8->2), per-lane scalar ----------------
    {
        float v1[10];
        #pragma unroll
        for (int i = 0; i < 10; ++i) v1[i] = ly[tid][i];

        float v2[8];
        #pragma unroll
        for (int o = 0; o < 8; ++o) {
            float s = fl2_b[o];
            #pragma unroll
            for (int i = 0; i < 10; ++i) s += v1[i] * fl2_w[o * 10 + i];
            v2[o] = fmaxf(s, 0.f);
        }

        float o0 = fl3_b[0], o1 = fl3_b[1];
        #pragma unroll
        for (int i = 0; i < 8; ++i) {
            o0 += v2[i] * fl3_w[i];
            o1 += v2[i] * fl3_w[8 + i];
        }

        if (active) {
            fx2 r; r.x = o0; r.y = o1;
            __builtin_nontemporal_store(r, (fx2*)(out + (size_t)b * 2));
        }
    }
}

extern "C" void kernel_launch(void* const* d_in, const int* in_sizes, int n_in,
                              void* d_out, int out_size, void* d_ws, size_t ws_size,
                              hipStream_t stream) {
    const float* nf     = (const float*)d_in[0];
    const float* adj    = (const float*)d_in[1];
    const float* proj_w = (const float*)d_in[2];
    const float* proj_b = (const float*)d_in[3];
    const float* fl1_w  = (const float*)d_in[4];
    const float* fl1_b  = (const float*)d_in[5];
    const float* fl2_w  = (const float*)d_in[6];
    const float* fl2_b  = (const float*)d_in[7];
    const float* fl3_w  = (const float*)d_in[8];
    const float* fl3_b  = (const float*)d_in[9];
    float* out = (float*)d_out;

    const int B    = in_sizes[0] / NN;       // node_features flat = B*18
    const int grid = (B + TPB - 1) / TPB;

    gcn_mlp_fused<<<grid, TPB, 0, stream>>>(nf, adj, proj_w, proj_b,
                                            fl1_w, fl1_b, fl2_w, fl2_b,
                                            fl3_w, fl3_b, out, B);
}
